// PointNetpp_23244363006052
// MI455X (gfx1250) — matrix operand order repaired and checked by reading, not compiled, not run on hardware
//
#include <hip/hip_runtime.h>
#include <hip/hip_bf16.h>
#include <stdint.h>

// ---------------------------------------------------------------------------
// PointNet++ forward for MI455X (gfx1250, wave32, WMMA).
// All MLP/FC layers run through v_wmma_f32_16x16x32_bf16 (bf16 in, f32 acc).
// Each wave computes a 16x(16*NT) output strip: A fragment loaded once per
// K-step, reused across NT B tiles (4x cut in dominant activation traffic).
// ---------------------------------------------------------------------------

typedef __attribute__((ext_vector_type(16))) __bf16 v16bf;
typedef __attribute__((ext_vector_type(8)))  float  v8f;
typedef __attribute__((ext_vector_type(4)))  unsigned int u32x4;

__device__ __forceinline__ unsigned short f2bf(float f) {
    unsigned int u = __float_as_uint(f);
    u += 0x7FFFu + ((u >> 16) & 1u);          // round-to-nearest-even
    return (unsigned short)(u >> 16);
}
__device__ __forceinline__ float bf2f(unsigned short h) {
    return __uint_as_float(((unsigned int)h) << 16);
}

// --------------------------- split (B,6,N) -> coords/norm (B,N,3) -----------
__global__ void k_split(const float* __restrict__ xyz,
                        float* __restrict__ coords, float* __restrict__ norm,
                        int B, int N) {
    int t = blockIdx.x * blockDim.x + threadIdx.x;
    if (t >= B * N) return;
    int b = t / N, n = t % N;
    const float* s = xyz + (size_t)b * 6 * N + n;
    float* c = coords + (size_t)t * 3;
    float* m = norm   + (size_t)t * 3;
    c[0] = s[0];             c[1] = s[(size_t)N];     c[2] = s[2*(size_t)N];
    m[0] = s[3*(size_t)N];   m[1] = s[4*(size_t)N];   m[2] = s[5*(size_t)N];
}

// --------------------------- farthest point sampling ------------------------
#define FPS_T 256
__global__ void k_fps(const float* __restrict__ xyz, int N, int npoint,
                      int* __restrict__ idx) {
    __shared__ float spts[4096 * 3];
    __shared__ float rv[FPS_T];
    __shared__ int   ri[FPS_T];
    __shared__ int   sfar;
    const int b = blockIdx.x;
    const int tid = threadIdx.x;
    for (int i = tid; i < N * 3; i += FPS_T) spts[i] = xyz[(size_t)b * N * 3 + i];
    float dist[16];
    const int ppt = (N + FPS_T - 1) / FPS_T;
    for (int k = 0; k < ppt; ++k) dist[k] = 1e10f;
    if (tid == 0) sfar = 0;
    __syncthreads();
    for (int it = 0; it < npoint; ++it) {
        int far = sfar;
        if (tid == 0) idx[(size_t)b * npoint + it] = far;
        float cx = spts[3*far], cy = spts[3*far+1], cz = spts[3*far+2];
        float best = -1.f; int bi = 0;
        for (int k = 0; k < ppt; ++k) {
            int i = k * FPS_T + tid;
            if (i < N) {
                float dx = spts[3*i]-cx, dy = spts[3*i+1]-cy, dz = spts[3*i+2]-cz;
                float nd = fminf(dist[k], dx*dx + dy*dy + dz*dz);
                dist[k] = nd;
                if (nd > best) { best = nd; bi = i; }
            }
        }
        rv[tid] = best; ri[tid] = bi;
        __syncthreads();
        for (int s = FPS_T / 2; s > 0; s >>= 1) {
            if (tid < s) {
                float ov = rv[tid+s]; int oi = ri[tid+s];
                if (ov > rv[tid] || (ov == rv[tid] && oi < ri[tid])) { rv[tid]=ov; ri[tid]=oi; }
            }
            __syncthreads();
        }
        if (tid == 0) sfar = ri[0];
        __syncthreads();
    }
}

// --------------------------- gather sampled centroids -----------------------
__global__ void k_gather_newxyz(const float* __restrict__ xyz, const int* __restrict__ fps,
                                float* __restrict__ nx, int B, int N, int S) {
    int t = blockIdx.x * blockDim.x + threadIdx.x;
    if (t >= B * S) return;
    int b = t / S;
    int g = fps[t];
    const float* p = xyz + ((size_t)b * N + g) * 3;
    nx[(size_t)t*3+0] = p[0]; nx[(size_t)t*3+1] = p[1]; nx[(size_t)t*3+2] = p[2];
}

// ---------- ball query + grouping into padded bf16 GEMM rows ----------------
// out row layout: [rel_xyz(3) | point feats(C) | zero pad .. Kpad)
__global__ void k_ball_group(const float* __restrict__ xyz, const float* __restrict__ nxyz,
                             const float* __restrict__ ptsF, const unsigned short* __restrict__ ptsB,
                             int C, int B, int N, int S, int K, float r2,
                             unsigned short* __restrict__ out, int Kpad) {
    int t = blockIdx.x * blockDim.x + threadIdx.x;
    if (t >= B * S) return;
    int b = t / S;
    float cx = nxyz[(size_t)t*3], cy = nxyz[(size_t)t*3+1], cz = nxyz[(size_t)t*3+2];
    int gi[64];
    int cnt = 0;
    for (int i = 0; i < N && cnt < K; ++i) {           // ascending index == first-K after sort
        const float* p = xyz + ((size_t)b * N + i) * 3;
        float dx = p[0]-cx, dy = p[1]-cy, dz = p[2]-cz;
        if (dx*dx + dy*dy + dz*dz <= r2) gi[cnt++] = i;
    }
    if (cnt == 0) gi[cnt++] = 0;
    for (int s = cnt; s < K; ++s) gi[s] = gi[0];
    for (int s = 0; s < K; ++s) {
        int g = gi[s];
        unsigned short* row = out + ((size_t)t * K + s) * Kpad;
        const float* p = xyz + ((size_t)b * N + g) * 3;
        row[0] = f2bf(p[0]-cx); row[1] = f2bf(p[1]-cy); row[2] = f2bf(p[2]-cz);
        if (ptsF) { for (int c = 0; c < C; ++c) row[3+c] = f2bf(ptsF[((size_t)b*N+g)*C + c]); }
        else      { for (int c = 0; c < C; ++c) row[3+c] = ptsB[((size_t)b*N+g)*C + c]; }
        for (int c = 3 + C; c < Kpad; ++c) row[c] = 0;
    }
}

// ---------- concat l2_xyz + l2_p -> padded bf16 rows (sa3 "group all") ------
__global__ void k_concat3(const float* __restrict__ l2xyz, const unsigned short* __restrict__ l2p,
                          unsigned short* __restrict__ out, int R, int C, int Kpad) {
    int t = blockIdx.x * blockDim.x + threadIdx.x;
    if (t >= R) return;
    unsigned short* row = out + (size_t)t * Kpad;
    row[0] = f2bf(l2xyz[(size_t)t*3]); row[1] = f2bf(l2xyz[(size_t)t*3+1]); row[2] = f2bf(l2xyz[(size_t)t*3+2]);
    for (int c = 0; c < C; ++c) row[3+c] = l2p[(size_t)t*C + c];
    for (int c = 3 + C; c < Kpad; ++c) row[c] = 0;
}

// --------------------------- max pool over samples --------------------------
__global__ void k_maxpool(const unsigned short* __restrict__ in, int R, int K, int C,
                          unsigned short* __restrict__ outB, float* __restrict__ outF) {
    int t = blockIdx.x * blockDim.x + threadIdx.x;
    if (t >= R * C) return;
    int r = t / C, c = t % C;
    const unsigned short* p = in + ((size_t)r * K) * C + c;
    float m = -3.4e38f;
    for (int k = 0; k < K; ++k) m = fmaxf(m, bf2f(p[(size_t)k * C]));
    if (outB) outB[t] = f2bf(m);
    if (outF) outF[t] = m;
}

// ---------- weight prep: f32 W[co][ci] -> bf16 W[co][Kpad] + BN fold --------
__global__ void k_prep(const float* __restrict__ W, const float* __restrict__ b,
                       const float* __restrict__ g, const float* __restrict__ be,
                       int CoR, int CoP, int Ci, int Kpad, float bnscale,
                       unsigned short* __restrict__ Wp,
                       float* __restrict__ scale, float* __restrict__ shift) {
    int t = blockIdx.x * blockDim.x + threadIdx.x;
    if (t >= CoP * Kpad) return;
    int co = t / Kpad, k = t % Kpad;
    Wp[t] = (co < CoR && k < Ci) ? f2bf(W[(size_t)co * Ci + k]) : (unsigned short)0;
    if (k == 0) {
        float s = 0.f, sh = 0.f;
        if (co < CoR) {
            if (g) { s = g[co] * bnscale; sh = b[co] * s + be[co]; }
            else   { s = 1.f;             sh = b[co]; }
        }
        scale[co] = s; shift[co] = sh;
    }
}

// ---------- WMMA linear: out = act( A[M,Kpad] * W[n][k]^T * scale + shift ) -
// One wave per 16x(16*NT) output strip.  flags: bit0 = relu, bit1 = f32 out.
// A fragment per K-step is loaded once and reused across the NT B tiles.
template <int NT>
__global__ void k_wmma_linear(const unsigned short* __restrict__ A, int lda,
                              const unsigned short* __restrict__ Wp,
                              const float* __restrict__ scale,
                              const float* __restrict__ shift,
                              unsigned short* __restrict__ outB,
                              float* __restrict__ outF,
                              int ldo, int M, int Kpad, int Nreal, int flags) {
    const int lane = threadIdx.x & 31;
    const int wv   = threadIdx.x >> 5;
    const int tm   = blockIdx.x * (blockDim.x >> 5) + wv;
    if (tm * 16 >= M) return;
    const int tn0   = blockIdx.y * NT;
    const int mrow  = tm * 16 + (lane & 15);
    const int khalf = (lane >> 4) * 8;          // A: lanes 16-31 hold K+8 halves
    const int nlane = lane & 15;
    const int koffB = (lane >> 4) * 8;          // B: lanes 16-31 hold K+8 halves (same as A)

    v8f acc[NT];
#pragma unroll
    for (int j = 0; j < NT; ++j) acc[j] = {};

    const unsigned short* ap = A + (size_t)mrow * lda + khalf;
    const unsigned short* wp[NT];
#pragma unroll
    for (int j = 0; j < NT; ++j)
        wp[j] = Wp + (size_t)((tn0 + j) * 16 + nlane) * Kpad + koffB;

    for (int k0 = 0; k0 < Kpad; k0 += 32) {
        union { u32x4 u[2]; v16bf v; } ua;
        ua.u[0] = *(const u32x4*)(ap);          // A lane: K {khalf..khalf+7}
        ua.u[1] = *(const u32x4*)(ap + 16);     //         K {khalf+16..khalf+23}
        __builtin_prefetch(ap + 32, 0, 1);
#pragma unroll
        for (int j = 0; j < NT; ++j) {
            union { u32x4 u[2]; v16bf v; } ub;
            ub.u[0] = *(const u32x4*)(wp[j]);   // B lane: K {koffB..+7}, {koffB+16..+23}
            ub.u[1] = *(const u32x4*)(wp[j] + 16);
            acc[j] = __builtin_amdgcn_wmma_f32_16x16x32_bf16(
                false, ua.v, false, ub.v, (short)0, acc[j], false, false);
            wp[j] += 32;
        }
        ap += 32;
    }
#pragma unroll
    for (int j = 0; j < NT; ++j) {              // C/D: VGPR i -> row i + 8*(lane>>4)
        const int ncol = (tn0 + j) * 16 + nlane;
        const bool nok = (ncol < Nreal);
        const float s  = nok ? scale[ncol] : 0.f;
        const float sh = nok ? shift[ncol] : 0.f;
#pragma unroll
        for (int i = 0; i < 8; ++i) {
            int gm = tm * 16 + i + 8 * (lane >> 4);
            float v = acc[j][i] * s + sh;
            if (flags & 1) v = fmaxf(v, 0.f);
            if (nok) {
                if (flags & 2) outF[(size_t)gm * ldo + ncol] = v;
                else           outB[(size_t)gm * ldo + ncol] = f2bf(v);
            }
        }
    }
}

// ---------------------------------------------------------------------------
struct Lw {
    const float *W, *b, *g, *be;
    int CoR, CoP, Ci, Kpad;
    unsigned short* Wp; float* sc; float* sh;
};

static void run_linear(const unsigned short* Ain, int lda, const Lw& L,
                       unsigned short* outB, float* outF, int ldo,
                       int M, int flags, hipStream_t s) {
    dim3 blk(128, 1, 1);                                   // 4 waves / block
    if (L.CoP % 64 == 0) {                                 // NT=4: 16x64 strip / wave
        dim3 grd((unsigned)((M / 16 + 3) / 4), (unsigned)(L.CoP / 64), 1);
        k_wmma_linear<4><<<grd, blk, 0, s>>>(Ain, lda, L.Wp, L.sc, L.sh,
                                             outB, outF, ldo, M, L.Kpad, L.CoR, flags);
    } else {                                               // NT=1 (fc3, CoP=16)
        dim3 grd((unsigned)((M / 16 + 3) / 4), (unsigned)(L.CoP / 16), 1);
        k_wmma_linear<1><<<grd, blk, 0, s>>>(Ain, lda, L.Wp, L.sc, L.sh,
                                             outB, outF, ldo, M, L.Kpad, L.CoR, flags);
    }
}

extern "C" void kernel_launch(void* const* d_in, const int* in_sizes, int n_in,
                              void* d_out, int out_size, void* d_ws, size_t ws_size,
                              hipStream_t stream) {
    (void)in_sizes; (void)n_in; (void)out_size; (void)ws_size;
    const int B = 16, N = 4096;
    const float BN = 0.9999950000374997f;                  // 1/sqrt(1+1e-5)
    const float* xyz = (const float*)d_in[0];
    float* out = (float*)d_out;                            // [0,160)=logits, [160,16544)=l3

    // Input leaf order: xyz, then params in setup_inputs() insertion order:
    // sa1 L0..L2 {W,b,g,be}, sa2 L0..L2, sa3 L0..L2, fc1W,fc1b,bn1g,bn1b,
    // fc2W,fc2b,bn2g,bn2b, fc3W,fc3b.
    const float* P[47];
    for (int i = 0; i < 47; ++i) P[i] = (const float*)d_in[i];

    Lw L[12] = {
        // sa1: 6->64->64->128   (Kpad 16,64,64)
        { P[1],  P[2],  P[3],  P[4],   64,   64,    6,   16, 0,0,0 },
        { P[5],  P[6],  P[7],  P[8],   64,   64,   64,   64, 0,0,0 },
        { P[9],  P[10], P[11], P[12], 128,  128,   64,   64, 0,0,0 },
        // sa2: 131->128->128->256 (Kpad 144,128,128)
        { P[13], P[14], P[15], P[16], 128,  128,  131,  144, 0,0,0 },
        { P[17], P[18], P[19], P[20], 128,  128,  128,  128, 0,0,0 },
        { P[21], P[22], P[23], P[24], 256,  256,  128,  128, 0,0,0 },
        // sa3: 259->256->512->1024 (Kpad 272,256,512)
        { P[25], P[26], P[27], P[28], 256,  256,  259,  272, 0,0,0 },
        { P[29], P[30], P[31], P[32], 512,  512,  256,  256, 0,0,0 },
        { P[33], P[34], P[35], P[36], 1024, 1024, 512,  512, 0,0,0 },
        // fc1 / fc2 (with BN), fc3 (plain bias, Co padded 10->16)
        { P[37], P[38], P[39], P[40], 512,  512, 1024, 1024, 0,0,0 },
        { P[41], P[42], P[43], P[44], 256,  256,  512,  512, 0,0,0 },
        { P[45], P[46], nullptr, nullptr, 10, 16, 256,  256, 0,0,0 },
    };

    // ---------------- workspace carve-up ----------------
    char* base = (char*)d_ws; size_t off = 0;
    auto carve = [&](size_t bytes) -> char* {
        char* p = base + off; off += (bytes + 255) & ~(size_t)255; return p;
    };
    float* coords = (float*)carve((size_t)B * N * 3 * 4);
    float* norm   = (float*)carve((size_t)B * N * 3 * 4);
    int*   fps1   = (int*)  carve((size_t)B * 512 * 4);
    float* l1xyz  = (float*)carve((size_t)B * 512 * 3 * 4);
    int*   fps2   = (int*)  carve((size_t)B * 128 * 4);
    float* l2xyz  = (float*)carve((size_t)B * 128 * 3 * 4);
    unsigned short* l1p  = (unsigned short*)carve((size_t)B * 512 * 128 * 2);
    unsigned short* l2p  = (unsigned short*)carve((size_t)B * 128 * 256 * 2);
    unsigned short* l3bf = (unsigned short*)carve((size_t)B * 1024 * 2);
    unsigned short* x1   = (unsigned short*)carve((size_t)B * 512 * 2);
    unsigned short* x2   = (unsigned short*)carve((size_t)B * 256 * 2);
    for (int i = 0; i < 12; ++i) {
        L[i].Wp = (unsigned short*)carve((size_t)L[i].CoP * L[i].Kpad * 2);
        L[i].sc = (float*)carve((size_t)L[i].CoP * 4);
        L[i].sh = (float*)carve((size_t)L[i].CoP * 4);
    }
    unsigned short* bufA = (unsigned short*)carve((size_t)18874368 * 2); // max: sa2 grouped 131072*144
    unsigned short* bufB = (unsigned short*)carve((size_t)33554432 * 2); // max: 262144*128 / 131072*256

    // ---------------- weight prep ----------------
    for (int i = 0; i < 12; ++i) {
        int tot = L[i].CoP * L[i].Kpad;
        k_prep<<<(tot + 255) / 256, 256, 0, stream>>>(
            L[i].W, L[i].b, L[i].g, L[i].be, L[i].CoR, L[i].CoP, L[i].Ci,
            L[i].Kpad, BN, L[i].Wp, L[i].sc, L[i].sh);
    }

    // ---------------- stage 0: split ----------------
    k_split<<<(B * N + 255) / 256, 256, 0, stream>>>(xyz, coords, norm, B, N);

    // ---------------- sa1: 4096 pts -> 512 centroids, r=0.2, K=32 ----------
    k_fps<<<B, FPS_T, 0, stream>>>(coords, N, 512, fps1);
    k_gather_newxyz<<<(B * 512 + 255) / 256, 256, 0, stream>>>(coords, fps1, l1xyz, B, N, 512);
    k_ball_group<<<(B * 512 + 255) / 256, 256, 0, stream>>>(
        coords, l1xyz, norm, nullptr, 3, B, N, 512, 32, 0.04f, bufA, 16);
    run_linear(bufA, 16, L[0], bufB, nullptr,  64, 262144, 1, stream);
    run_linear(bufB, 64, L[1], bufA, nullptr,  64, 262144, 1, stream);
    run_linear(bufA, 64, L[2], bufB, nullptr, 128, 262144, 1, stream);
    k_maxpool<<<(8192 * 128 + 255) / 256, 256, 0, stream>>>(bufB, 8192, 32, 128, l1p, nullptr);

    // ---------------- sa2: 512 -> 128 centroids, r=0.4, K=64 ---------------
    k_fps<<<B, FPS_T, 0, stream>>>(l1xyz, 512, 128, fps2);
    k_gather_newxyz<<<(B * 128 + 255) / 256, 256, 0, stream>>>(l1xyz, fps2, l2xyz, B, 512, 128);
    k_ball_group<<<(B * 128 + 255) / 256, 256, 0, stream>>>(
        l1xyz, l2xyz, nullptr, l1p, 128, B, 512, 128, 64, 0.16f, bufA, 144);
    run_linear(bufA, 144, L[3], bufB, nullptr, 128, 131072, 1, stream);
    run_linear(bufB, 128, L[4], bufA, nullptr, 128, 131072, 1, stream);
    run_linear(bufA, 128, L[5], bufB, nullptr, 256, 131072, 1, stream);
    k_maxpool<<<(2048 * 256 + 255) / 256, 256, 0, stream>>>(bufB, 2048, 64, 256, l2p, nullptr);

    // ---------------- sa3: group-all over 128 points ------------------------
    k_concat3<<<(2048 + 255) / 256, 256, 0, stream>>>(l2xyz, l2p, bufA, 2048, 256, 272);
    run_linear(bufA, 272, L[6], bufB, nullptr,  256, 2048, 1, stream);
    run_linear(bufB, 256, L[7], bufA, nullptr,  512, 2048, 1, stream);
    run_linear(bufA, 512, L[8], bufB, nullptr, 1024, 2048, 1, stream);
    // l3: f32 to d_out (second output) + bf16 for FC head
    k_maxpool<<<(16 * 1024 + 255) / 256, 256, 0, stream>>>(bufB, 16, 128, 1024, l3bf, out + 160);

    // ---------------- FC head ----------------
    run_linear(l3bf, 1024, L[9],  x1, nullptr, 512, 16, 1, stream);
    run_linear(x1,    512, L[10], x2, nullptr, 256, 16, 1, stream);
    run_linear(x2,    256, L[11], nullptr, out, 10, 16, 2, stream);   // f32 logits, no relu
}
